// SimFCOSHead_40699110097455
// MI455X (gfx1250) — matrix operand order repaired and checked by reading, not compiled, not run on hardware
//
#include <hip/hip_runtime.h>

// ---------------------------------------------------------------------------
// CDNA5 / gfx1250 FCOS head (deformable conv towers).
// All convs are implicit GEMMs on V_WMMA_F32_16X16X32_BF16 with 4x2 register
// blocking per wave; B matrix stored transposed ([pixel][K]) so fragments are
// contiguous 32-byte vector loads; A keeps natural K order at repack time.
// ---------------------------------------------------------------------------

typedef __bf16 bf16_t;
typedef __attribute__((ext_vector_type(16))) __bf16 v16bf;
typedef __attribute__((ext_vector_type(8)))  float  v8f;

#define C_IN   256
#define BATCH  2
#define NGRP   32

static __device__ __forceinline__ int imin(int a, int b) { return a < b ? a : b; }
static __device__ __forceinline__ int imax(int a, int b) { return a > b ? a : b; }

// ---------------------------------------------------------------------------
// Weight repack: fp32 [M][K] -> bf16 [Mp64][K], zero pad rows M..Mp64-1.
// Within each 32-wide K block, K order is kept natural (contiguous order)
// so a lane's A fragment is 16 contiguous bf16 (k = 16*h + s), matching B,
// at offset 16*h inside the block.
// ---------------------------------------------------------------------------
__global__ void repack_w(const float* __restrict__ w, bf16_t* __restrict__ wp,
                         int M, int Mp64, int K) {
    int t = blockIdx.x * 256 + threadIdx.x;
    if (t >= Mp64 * K) return;
    int m = t / K;
    int k = t - m * K;
    int kk = k & 31;
    int grp = (kk >> 3) & 3;
    int kk2 = kk + ((grp == 1) ? 0 : (grp == 2) ? 0 : 0);
    int kdst = (k & ~31) + kk2;
    wp[(long)m * K + kdst] = (m < M) ? (bf16_t)w[(long)m * K + k] : (bf16_t)0.0f;
}

// ---------------------------------------------------------------------------
// im2col (3x3 pad 1) into transposed patches: Bt bf16 [B][np_pad][C*9]
// k order = ci*9 + (ky*3+kx); pixel-major so writes are fully coalesced.
// ---------------------------------------------------------------------------
__global__ void im2col3x3(const float* __restrict__ x, bf16_t* __restrict__ Bt,
                          int C, int H, int W, int np_pad) {
    int np = H * W;
    int K = C * 9;
    int t = blockIdx.x * 256 + threadIdx.x;
    if (t >= np * K) return;
    int b = blockIdx.z;
    int p = t / K;
    int k = t - p * K;
    int ci = k / 9;
    int j  = k - ci * 9;
    int y  = p / W;
    int xx = p - y * W;
    int sy = y + j / 3 - 1;
    int sx = xx + (j % 3) - 1;
    float v = 0.0f;
    if (sy >= 0 && sy < H && sx >= 0 && sx < W)
        v = x[(((long)b * C + ci) * H + sy) * W + sx];
    Bt[((long)b * np_pad + p) * K + k] = (bf16_t)v;
}

// ---------------------------------------------------------------------------
// fp32 [B][256][np] -> bf16 transposed [B][np_pad][256] (for the 1x1 conv)
// ---------------------------------------------------------------------------
__global__ void transpose_bf16(const float* __restrict__ in, bf16_t* __restrict__ Bt,
                               int np, int np_pad) {
    int t = blockIdx.x * 256 + threadIdx.x;
    if (t >= np * C_IN) return;
    int b = blockIdx.z;
    int p = t / C_IN;
    int c = t - p * C_IN;
    Bt[((long)b * np_pad + p) * C_IN + c] = (bf16_t)in[((long)b * C_IN + c) * np + p];
}

// ---------------------------------------------------------------------------
// Deformable sampling into transposed patches Bt bf16 [B][np_pad][C*9].
// om fp32 [B][27][np]: (2j)=offy, (2j+1)=offx, (18+j)=mask logits.
// Per-corner validity identical to the reference.
// ---------------------------------------------------------------------------
__global__ void deform_gather(const float* __restrict__ x, const float* __restrict__ om,
                              bf16_t* __restrict__ Bt, int H, int W, int np_pad) {
    int np = H * W;
    int t = blockIdx.x * 256 + threadIdx.x;
    if (t >= 9 * np) return;
    int b = blockIdx.z;
    int j = t / np;
    int p = t - j * np;
    int y = p / W;
    int xx = p - y * W;

    const float* omb = om + (long)b * 27 * np;
    float offy = omb[(long)(2 * j) * np + p];
    float offx = omb[(long)(2 * j + 1) * np + p];
    float mlog = omb[(long)(18 + j) * np + p];
    float mk = 1.0f / (1.0f + expf(-mlog));

    float ys = (float)(y + j / 3 - 1) + offy;
    float xs = (float)(xx + (j % 3) - 1) + offx;
    float y0f = floorf(ys), x0f = floorf(xs);
    float wy = ys - y0f, wx = xs - x0f;
    int y0 = (int)y0f, x0 = (int)x0f;
    int y1 = y0 + 1,   x1 = x0 + 1;

    float v00 = (1.0f - wy) * (1.0f - wx) * ((y0 >= 0 && y0 < H && x0 >= 0 && x0 < W) ? mk : 0.0f);
    float v01 = (1.0f - wy) * wx          * ((y0 >= 0 && y0 < H && x1 >= 0 && x1 < W) ? mk : 0.0f);
    float v10 = wy * (1.0f - wx)          * ((y1 >= 0 && y1 < H && x0 >= 0 && x0 < W) ? mk : 0.0f);
    float v11 = wy * wx                   * ((y1 >= 0 && y1 < H && x1 >= 0 && x1 < W) ? mk : 0.0f);

    int yc0 = imin(imax(y0, 0), H - 1), yc1 = imin(imax(y1, 0), H - 1);
    int xc0 = imin(imax(x0, 0), W - 1), xc1 = imin(imax(x1, 0), W - 1);
    long i00 = (long)yc0 * W + xc0, i01 = (long)yc0 * W + xc1;
    long i10 = (long)yc1 * W + xc0, i11 = (long)yc1 * W + xc1;

    const float* xb = x + (long)b * C_IN * np;
    bf16_t* row = Bt + ((long)b * np_pad + p) * (C_IN * 9) + j;
    for (int ci = 0; ci < C_IN; ++ci) {
        const float* xc = xb + (long)ci * np;
        float val = v00 * xc[i00] + v01 * xc[i01] + v10 * xc[i10] + v11 * xc[i11];
        row[(long)ci * 9] = (bf16_t)val;
    }
}

// ---------------------------------------------------------------------------
// GroupNorm (32 groups, eps 1e-5) + ReLU. One block per (group, batch).
// ---------------------------------------------------------------------------
__global__ __launch_bounds__(256) void groupnorm_relu(
    const float* __restrict__ x, const float* __restrict__ gam,
    const float* __restrict__ bet, float* __restrict__ out, int np) {
    __shared__ float s1[256], s2[256];
    __shared__ float s_mu, s_rstd;
    int g = blockIdx.x;
    int b = blockIdx.z;
    int tid = threadIdx.x;
    const int cpg = C_IN / NGRP;  // 8
    long base = ((long)b * C_IN + (long)g * cpg) * np;
    int cnt = cpg * np;

    float sum = 0.0f, sq = 0.0f;
    for (int i = tid; i < cnt; i += 256) {
        float v = x[base + i];
        sum += v; sq += v * v;
    }
    s1[tid] = sum; s2[tid] = sq;
    __syncthreads();
    for (int s = 128; s > 0; s >>= 1) {
        if (tid < s) { s1[tid] += s1[tid + s]; s2[tid] += s2[tid + s]; }
        __syncthreads();
    }
    if (tid == 0) {
        float mu = s1[0] / (float)cnt;
        float var = s2[0] / (float)cnt - mu * mu;
        s_mu = mu;
        s_rstd = rsqrtf(var + 1e-5f);
    }
    __syncthreads();
    float mu = s_mu, rstd = s_rstd;
    for (int i = tid; i < cnt; i += 256) {
        int c = g * cpg + i / np;
        float v = (x[base + i] - mu) * rstd * gam[c] + bet[c];
        out[base + i] = fmaxf(v, 0.0f);
    }
}

// ---------------------------------------------------------------------------
// Implicit-GEMM conv, 64x32 tile per wave (4x2 WMMA register blocking).
//   A  : bf16 [Mp64][K]     natural-K-order packed weights
//   Bt : bf16 [B][np_pad][K] transposed patches
//   out: fp32 [B][M][np]
// Per K-step: 4 A-frag + 2 B-frag contiguous 32B loads, 8 WMMAs.
// Epilogue: +bias, *scale, optional ReLU, bounds-guarded stores.
// ---------------------------------------------------------------------------
__global__ __launch_bounds__(32) void gemm_wmma_bf16(
    const bf16_t* __restrict__ A, const bf16_t* __restrict__ Bt,
    const float* __restrict__ bias, float* __restrict__ out,
    int M, int K, int np, int np_pad,
    const float* __restrict__ scale_ptr, int scale_idx, int do_relu) {
    const int lane = threadIdx.x;       // wave32
    const int n = lane & 15;
    const int h = lane >> 4;
    const int nbase = blockIdx.x * 32;
    const int mbase = blockIdx.y * 64;
    const int b = blockIdx.z;

    const bf16_t* arow0 = A + (long)(mbase +  0 + n) * K;
    const bf16_t* arow1 = A + (long)(mbase + 16 + n) * K;
    const bf16_t* arow2 = A + (long)(mbase + 32 + n) * K;
    const bf16_t* arow3 = A + (long)(mbase + 48 + n) * K;
    const bf16_t* brow0 = Bt + ((long)b * np_pad + nbase +  0 + n) * K;
    const bf16_t* brow1 = Bt + ((long)b * np_pad + nbase + 16 + n) * K;

    v8f acc00 = {}, acc01 = {}, acc10 = {}, acc11 = {};
    v8f acc20 = {}, acc21 = {}, acc30 = {}, acc31 = {};

    for (int kb = 0; kb < K; kb += 32) {
        int ko = kb + 16 * h;
        v16bf b0 = *(const v16bf*)(brow0 + ko);
        v16bf b1 = *(const v16bf*)(brow1 + ko);
        v16bf a0 = *(const v16bf*)(arow0 + ko);
        v16bf a1 = *(const v16bf*)(arow1 + ko);
        v16bf a2 = *(const v16bf*)(arow2 + ko);
        v16bf a3 = *(const v16bf*)(arow3 + ko);
        acc00 = __builtin_amdgcn_wmma_f32_16x16x32_bf16(false, a0, false, b0, (short)0, acc00, false, false);
        acc01 = __builtin_amdgcn_wmma_f32_16x16x32_bf16(false, a0, false, b1, (short)0, acc01, false, false);
        acc10 = __builtin_amdgcn_wmma_f32_16x16x32_bf16(false, a1, false, b0, (short)0, acc10, false, false);
        acc11 = __builtin_amdgcn_wmma_f32_16x16x32_bf16(false, a1, false, b1, (short)0, acc11, false, false);
        acc20 = __builtin_amdgcn_wmma_f32_16x16x32_bf16(false, a2, false, b0, (short)0, acc20, false, false);
        acc21 = __builtin_amdgcn_wmma_f32_16x16x32_bf16(false, a2, false, b1, (short)0, acc21, false, false);
        acc30 = __builtin_amdgcn_wmma_f32_16x16x32_bf16(false, a3, false, b0, (short)0, acc30, false, false);
        acc31 = __builtin_amdgcn_wmma_f32_16x16x32_bf16(false, a3, false, b1, (short)0, acc31, false, false);
    }

    float s = scale_ptr ? scale_ptr[scale_idx] : 1.0f;
    const v8f* accs[4][2] = {{&acc00, &acc01}, {&acc10, &acc11},
                             {&acc20, &acc21}, {&acc30, &acc31}};
    #pragma unroll
    for (int mt = 0; mt < 4; ++mt) {
        #pragma unroll
        for (int nt = 0; nt < 2; ++nt) {
            int pix = nbase + 16 * nt + n;
            #pragma unroll
            for (int r = 0; r < 8; ++r) {
                int m = mbase + 16 * mt + r + 8 * h;
                if (m < M && pix < np) {
                    float v = (*accs[mt][nt])[r];
                    if (bias) v += bias[m];
                    v *= s;
                    if (do_relu) v = fmaxf(v, 0.0f);
                    out[((long)b * M + m) * np + pix] = v;
                }
            }
        }
    }
}

// ---------------------------------------------------------------------------
// Host orchestration.
// Input order (sorted-key pytree flatten):
// 0..4: p3..p7
// 5: bbox_pred.b  6: bbox_pred.w
// box tower: 7 conv2.b, 8 conv2.w, 9 conv4.b, 10 conv4.w, 11 dcn1.b_off,
//            12 dcn1.w, 13 dcn1.w_off, 14 dcn3.b_off, 15 dcn3.w, 16 dcn3.w_off,
//            17 gn1.b, 18 gn1.g, 19 gn2.b, 20 gn2.g, 21 gn3.b, 22 gn3.g
// cls tower: 23..38 same field order
// 39 cls_logits.b, 40 cls_logits.w, 41 ctrness.b, 42 ctrness.w, 43 scales
// ---------------------------------------------------------------------------
struct TIdx {
    int conv2b, conv2w, conv4b, conv4w;
    int d1boff, d1w, d1woff, d3boff, d3w, d3woff;
    int gn1b, gn1g, gn2b, gn2g, gn3b, gn3g;
};

extern "C" void kernel_launch(void* const* d_in, const int* in_sizes, int n_in,
                              void* d_out, int out_size, void* d_ws, size_t ws_size,
                              hipStream_t stream) {
    (void)in_sizes; (void)n_in; (void)out_size; (void)ws_size;
    const int Hs[5] = {100, 50, 25, 13, 7};
    const int NPMAX_PAD = 10016;  // 100*100 rounded up to 32
    const int K33 = C_IN * 9;     // 2304

    const TIdx TI[2] = {
        {23, 24, 25, 26, 27, 28, 29, 30, 31, 32, 33, 34, 35, 36, 37, 38},  // cls
        {7, 8, 9, 10, 11, 12, 13, 14, 15, 16, 17, 18, 19, 20, 21, 22},     // box
    };

    // ---- workspace carve-up -------------------------------------------------
    char* ws = (char*)d_ws;
    size_t off = 0;
    auto alloc = [&](size_t bytes) -> char* {
        char* p = ws + off;
        off = (off + bytes + 255) & ~(size_t)255;
        return p;
    };
    bf16_t* Bt = (bf16_t*)alloc((size_t)BATCH * NPMAX_PAD * K33 * sizeof(bf16_t));
    float*  t0 = (float*)alloc((size_t)BATCH * C_IN * 10000 * sizeof(float));
    float*  t1 = (float*)alloc((size_t)BATCH * C_IN * 10000 * sizeof(float));
    float*  om = (float*)alloc((size_t)BATCH * 27 * 10000 * sizeof(float));

    bf16_t *wp_off1[2], *wp_dcn1[2], *wp_conv2[2], *wp_off3[2], *wp_dcn3[2], *wp_conv4[2];
    for (int t = 0; t < 2; ++t) {
        wp_off1[t]  = (bf16_t*)alloc((size_t)64 * K33 * sizeof(bf16_t));
        wp_dcn1[t]  = (bf16_t*)alloc((size_t)256 * K33 * sizeof(bf16_t));
        wp_conv2[t] = (bf16_t*)alloc((size_t)256 * K33 * sizeof(bf16_t));
        wp_off3[t]  = (bf16_t*)alloc((size_t)64 * K33 * sizeof(bf16_t));
        wp_dcn3[t]  = (bf16_t*)alloc((size_t)256 * K33 * sizeof(bf16_t));
        wp_conv4[t] = (bf16_t*)alloc((size_t)256 * 256 * sizeof(bf16_t));
    }
    bf16_t* wp_cls = (bf16_t*)alloc((size_t)128 * K33 * sizeof(bf16_t));
    bf16_t* wp_box = (bf16_t*)alloc((size_t)64 * K33 * sizeof(bf16_t));
    bf16_t* wp_ctr = (bf16_t*)alloc((size_t)64 * K33 * sizeof(bf16_t));

    // ---- helper launchers ---------------------------------------------------
    auto repack = [&](int widx, bf16_t* dst, int M, int Mp64, int K) {
        int tot = Mp64 * K;
        repack_w<<<(tot + 255) / 256, 256, 0, stream>>>((const float*)d_in[widx], dst, M, Mp64, K);
    };
    auto gemm = [&](const bf16_t* A, const float* bias, float* o,
                    int M, int Mp64, int K, int np, int np_pad,
                    const float* scl, int sidx, int relu) {
        dim3 grid((np + 31) / 32, Mp64 / 64, BATCH);
        gemm_wmma_bf16<<<grid, 32, 0, stream>>>(A, Bt, bias, o, M, K, np, np_pad, scl, sidx, relu);
    };
    auto im2col = [&](const float* x, int H, int W, int np_pad) {
        int tot = C_IN * 9 * H * W;
        dim3 grid((tot + 255) / 256, 1, BATCH);
        im2col3x3<<<grid, 256, 0, stream>>>(x, Bt, C_IN, H, W, np_pad);
    };
    auto deform = [&](const float* x, int H, int W, int np_pad) {
        int tot = 9 * H * W;
        dim3 grid((tot + 255) / 256, 1, BATCH);
        deform_gather<<<grid, 256, 0, stream>>>(x, om, Bt, H, W, np_pad);
    };
    auto gnrelu = [&](const float* x, int gidx, int bidx, float* o, int np) {
        dim3 grid(NGRP, 1, BATCH);
        groupnorm_relu<<<grid, 256, 0, stream>>>(x, (const float*)d_in[gidx],
                                                 (const float*)d_in[bidx], o, np);
    };
    auto cvt = [&](const float* x, int np, int np_pad) {
        int tot = np * C_IN;
        dim3 grid((tot + 255) / 256, 1, BATCH);
        transpose_bf16<<<grid, 256, 0, stream>>>(x, Bt, np, np_pad);
    };

    // ---- repack all weights (bf16, padded, natural K order) ---------------
    for (int t = 0; t < 2; ++t) {
        repack(TI[t].d1woff, wp_off1[t], 27, 64, K33);
        repack(TI[t].d1w,    wp_dcn1[t], 256, 256, K33);
        repack(TI[t].conv2w, wp_conv2[t], 256, 256, K33);
        repack(TI[t].d3woff, wp_off3[t], 27, 64, K33);
        repack(TI[t].d3w,    wp_dcn3[t], 256, 256, K33);
        repack(TI[t].conv4w, wp_conv4[t], 256, 256, 256);
    }
    repack(40, wp_cls, 80, 128, K33);
    repack(6,  wp_box, 4, 64, K33);
    repack(42, wp_ctr, 1, 64, K33);

    // ---- tower runner -------------------------------------------------------
    auto run_tower = [&](int t, const float* xin, int H, int W, int np_pad) -> float* {
        int np = H * W;
        // dcn1
        im2col(xin, H, W, np_pad);
        gemm(wp_off1[t], (const float*)d_in[TI[t].d1boff], om, 27, 64, K33, np, np_pad, nullptr, 0, 0);
        deform(xin, H, W, np_pad);
        gemm(wp_dcn1[t], nullptr, t0, 256, 256, K33, np, np_pad, nullptr, 0, 0);
        gnrelu(t0, TI[t].gn1g, TI[t].gn1b, t1, np);
        // conv2
        im2col(t1, H, W, np_pad);
        gemm(wp_conv2[t], (const float*)d_in[TI[t].conv2b], t0, 256, 256, K33, np, np_pad, nullptr, 0, 0);
        gnrelu(t0, TI[t].gn2g, TI[t].gn2b, t1, np);
        // dcn3
        im2col(t1, H, W, np_pad);
        gemm(wp_off3[t], (const float*)d_in[TI[t].d3boff], om, 27, 64, K33, np, np_pad, nullptr, 0, 0);
        deform(t1, H, W, np_pad);
        gemm(wp_dcn3[t], nullptr, t0, 256, 256, K33, np, np_pad, nullptr, 0, 0);
        gnrelu(t0, TI[t].gn3g, TI[t].gn3b, t1, np);
        // conv4 (1x1)
        cvt(t1, np, np_pad);
        gemm(wp_conv4[t], (const float*)d_in[TI[t].conv4b], t0, 256, 256, 256, np, np_pad, nullptr, 0, 0);
        return t0;
    };

    // ---- output layout: cls p3..p7 | box p3..p7 | ctr p3..p7 ----------------
    long sumnp = 0;
    for (int i = 0; i < 5; ++i) sumnp += (long)Hs[i] * Hs[i];
    float* out = (float*)d_out;
    long clsOff = 0;
    long boxOff = (long)BATCH * 80 * sumnp;
    long ctrOff = boxOff + (long)BATCH * 4 * sumnp;
    const float* scales = (const float*)d_in[43];

    for (int i = 0; i < 5; ++i) {
        int H = Hs[i], W = Hs[i], np = H * W;
        int np_pad = ((np + 31) / 32) * 32;
        const float* feat = (const float*)d_in[i];

        // cls tower + cls_logits + ctrness
        float* cf = run_tower(0, feat, H, W, np_pad);
        im2col(cf, H, W, np_pad);
        gemm(wp_cls, (const float*)d_in[39], out + clsOff, 80, 128, K33, np, np_pad, nullptr, 0, 0);
        gemm(wp_ctr, (const float*)d_in[41], out + ctrOff, 1, 64, K33, np, np_pad, nullptr, 0, 0);

        // box tower + bbox_pred (scaled + ReLU)
        float* bx = run_tower(1, feat, H, W, np_pad);
        im2col(bx, H, W, np_pad);
        gemm(wp_box, (const float*)d_in[5], out + boxOff, 4, 64, K33, np, np_pad, scales, i, 1);

        clsOff += (long)BATCH * 80 * np;
        boxOff += (long)BATCH * 4 * np;
        ctrOff += (long)BATCH * 1 * np;
    }
}
